// DinoGazeSpade_1047972020904
// MI455X (gfx1250) — matrix operand order repaired and checked by reading, not compiled, not run on hardware
//
#include <hip/hip_runtime.h>

// ---------------- problem constants ----------------
#define SEGS   64
#define BATCH  2
#define HIMG   336
#define HP     24
#define CD     768
#define CM     3072
#define RH     48
#define HID    128
#define HW     2304   // 48*48
#define HALOW  50     // 48 + 1px zero halo each side
#define HALOPIX 2500  // 50*50
#define LNEPS  1e-12f

typedef __attribute__((ext_vector_type(16))) __bf16 v16bf;
typedef __attribute__((ext_vector_type(8)))  __bf16 v8bf;
typedef __attribute__((ext_vector_type(8)))  float  v8f;

__device__ __forceinline__ unsigned short f2bf(float f) {
  unsigned int u = __float_as_uint(f);
  u += 0x7FFFu + ((u >> 16) & 1u);           // round to nearest even
  return (unsigned short)(u >> 16);
}

union ABFrag { v16bf v; v8bf h[2]; unsigned int u[8]; };

#define WMMA_BF16(A_, B_, C_) \
  __builtin_amdgcn_wmma_f32_16x16x32_bf16(false, (A_), false, (B_), (short)0, (C_), false, false)

// ---------------- buffer zero (for halo borders; full-buffer for simplicity) ----------------
__global__ void zero_u32(unsigned int* __restrict__ p, int n) {
  int i = blockIdx.x * blockDim.x + threadIdx.x;
  if (i < n) p[i] = 0u;
}

// ---------------- weight repack: OIHW fp32 -> [tap][Opad][I] bf16 ----------------
__global__ void repack_w(const float* __restrict__ w, unsigned short* __restrict__ out,
                         int O, int I, int Opad, int total) {
  int idx = blockIdx.x * blockDim.x + threadIdx.x;
  if (idx >= total) return;
  int i = idx % I;
  int o = (idx / I) % Opad;
  int t = idx / (I * Opad);
  float v = (o < O) ? w[((size_t)o * I + i) * 9 + t] : 0.0f;
  out[idx] = f2bf(v);
}

// ---------------- segment means (deterministic, LDS) -> meansT bf16 [b][CD][SEGS] ----------------
__global__ void compute_meansT(const float* __restrict__ f_sem,
                               const int* __restrict__ segmap,
                               unsigned short* __restrict__ meansT) {
  __shared__ int   ids[576];
  __shared__ float cnt[SEGS];
  __shared__ float acc[SEGS * 128];
  const int b   = blockIdx.x;
  const int tid = threadIdx.x;    // 128 threads
  for (int p = tid; p < 576; p += 128) {
    int py = p / HP, px = p % HP;
    int sv = segmap[(size_t)b * HIMG * HIMG + (py * 14) * HIMG + px * 14];
    ids[p] = min(max(sv, 0), SEGS - 1);
  }
  __syncthreads();
  if (tid < SEGS) {
    float c = 0.0f;
    for (int p = 0; p < 576; ++p) c += (ids[p] == tid) ? 1.0f : 0.0f;
    cnt[tid] = c;
  }
  __syncthreads();
  for (int chunk = 0; chunk < 6; ++chunk) {          // 6 chunks of 128 channels
    for (int e = tid; e < SEGS * 128; e += 128) acc[e] = 0.0f;
    __syncthreads();
    const float* fb = f_sem + ((size_t)b * CD + chunk * 128) * 576;
    for (int p = 0; p < 576; ++p) {                  // deterministic order
      int s = ids[p];
      acc[s * 128 + tid] += fb[(size_t)tid * 576 + p];
    }
    __syncthreads();
    for (int e = tid; e < SEGS * 128; e += 128) {
      int s = e >> 7, c = e & 127;
      float cn = cnt[s];
      float m  = (cn > 0.0f) ? acc[e] / cn : 0.0f;
      meansT[((size_t)b * CD + chunk * 128 + c) * SEGS + s] = f2bf(m);
    }
    __syncthreads();
  }
}

// ---------------- Wseg: 13-tap triangle downsample weights over seg ids -> bf16 [b][HW][SEGS] ----
__global__ void compute_wseg(const int* __restrict__ segmap,
                             unsigned short* __restrict__ wseg) {
  __shared__ float hist[64 * 65];                    // pad 65 to avoid bank conflicts
  const int tid = threadIdx.x;                       // 64 threads
  const int b   = blockIdx.x / 36;
  const int pix = (blockIdx.x % 36) * 64 + tid;
  float* h = hist + tid * 65;
  for (int s = 0; s < SEGS; ++s) h[s] = 0.0f;
  const int oy = pix / RH, ox = pix % RH;
  const int cy = 7 * oy + 3, cx = 7 * ox + 3;        // source center of 7x downsample
  float wsum = 0.0f;
  const int* sm = segmap + (size_t)b * HIMG * HIMG;
  for (int dy = -6; dy <= 6; ++dy) {
    int iy = cy + dy;
    if (iy < 0 || iy >= HIMG) continue;
    float wy = (float)(7 - (dy < 0 ? -dy : dy));
    for (int dx = -6; dx <= 6; ++dx) {
      int ix = cx + dx;
      if (ix < 0 || ix >= HIMG) continue;
      float wx = (float)(7 - (dx < 0 ? -dx : dx));
      int sv = sm[iy * HIMG + ix];
      sv = min(max(sv, 0), SEGS - 1);
      float w = wy * wx;
      h[sv] += w;
      wsum  += w;
    }
  }
  float inv = 1.0f / wsum;
  unsigned short* o = wseg + ((size_t)b * HW + pix) * SEGS;
  for (int s = 0; s < SEGS; ++s) o[s] = f2bf(h[s] * inv);
}

// ---------------- LayerNorm stats (two-pass, deterministic) ----------------
__global__ void stats_partial(const float* __restrict__ x, float* __restrict__ part,
                              int n_per_batch, int nblk) {
  __shared__ float l1[256], l2[256];
  const int b   = blockIdx.x / nblk;
  const int blk = blockIdx.x % nblk;
  const float* xb = x + (size_t)b * n_per_batch;
  int chunk = (n_per_batch + nblk - 1) / nblk;
  int s = blk * chunk;
  int e = min(s + chunk, n_per_batch);
  float s1 = 0.0f, s2 = 0.0f;
  for (int i = s + threadIdx.x; i < e; i += 256) { float v = xb[i]; s1 += v; s2 += v * v; }
  l1[threadIdx.x] = s1; l2[threadIdx.x] = s2;
  __syncthreads();
  for (int o = 128; o > 0; o >>= 1) {
    if (threadIdx.x < o) { l1[threadIdx.x] += l1[threadIdx.x + o]; l2[threadIdx.x] += l2[threadIdx.x + o]; }
    __syncthreads();
  }
  if (threadIdx.x == 0) {
    part[((size_t)b * nblk + blk) * 2 + 0] = l1[0];
    part[((size_t)b * nblk + blk) * 2 + 1] = l2[0];
  }
}

__global__ void stats_final(const float* __restrict__ part, float* __restrict__ stats,
                            int nblk, float inv_n) {
  __shared__ float l1[64], l2[64];
  const int b = blockIdx.x;
  float s1 = 0.0f, s2 = 0.0f;
  for (int i = threadIdx.x; i < nblk; i += 64) {
    s1 += part[((size_t)b * nblk + i) * 2 + 0];
    s2 += part[((size_t)b * nblk + i) * 2 + 1];
  }
  l1[threadIdx.x] = s1; l2[threadIdx.x] = s2;
  __syncthreads();
  for (int o = 32; o > 0; o >>= 1) {
    if (threadIdx.x < o) { l1[threadIdx.x] += l1[threadIdx.x + o]; l2[threadIdx.x] += l2[threadIdx.x + o]; }
    __syncthreads();
  }
  if (threadIdx.x == 0) {
    float mean = l1[0] * inv_n;
    stats[b * 2 + 0] = mean;
    stats[b * 2 + 1] = l2[0] * inv_n - mean * mean;
  }
}

// ---------------- WMMA implicit-GEMM conv, 2Mx2N register blocking ----------------
// A bf16: halo layout [b][HALOPIX][Cin] (in_halo=1) or flat [b][HW][Cin] (in_halo=0, taps must be 1)
// Wt bf16 [tap][CoutPad][Cin] (+ optional batch stride). Output bf16 halo layout [b][HALOPIX][CoutPad].
// Wave computes a 32(M)x32(N) tile: 4 accumulators, A-pair x B-pair -> 4 WMMA per k-step.
__global__ void wmma_conv_bf16(const unsigned short* __restrict__ A,
                               const unsigned short* __restrict__ Wt,
                               const float* __restrict__ bias,
                               unsigned short* __restrict__ out,
                               int Cin, int CoutPad, int taps,
                               long long wt_bstride, int relu, int in_halo) {
  const int lane = threadIdx.x & 31;
  const int wave = threadIdx.x >> 5;
  const int n0   = blockIdx.y * 128 + wave * 32;
  if (n0 >= CoutPad) return;
  const int m0   = blockIdx.x * 32;
  const int b    = blockIdx.z;
  const int rsel = lane >> 4;
  const int lmod = lane & 15;
  const int pix0 = m0 + lmod, pix1 = pix0 + 16;
  const int oy0 = pix0 / RH, ox0 = pix0 % RH;
  const int oy1 = pix1 / RH, ox1 = pix1 % RH;
  const int apix = in_halo ? HALOPIX : HW;
  const unsigned short* Ab = A  + (size_t)b * apix * Cin;
  const unsigned short* Wb = Wt + (size_t)b * wt_bstride;
  v8f acc[2][2];
#pragma unroll
  for (int m = 0; m < 2; ++m)
#pragma unroll
    for (int n = 0; n < 2; ++n)
#pragma unroll
      for (int i = 0; i < 8; ++i) acc[m][n][i] = 0.0f;

  for (int t = 0; t < taps; ++t) {
    const int ty = (taps == 1) ? 0 : (t / 3 - 1);
    const int tx = (taps == 1) ? 0 : (t % 3 - 1);
    const int s0 = in_halo ? ((oy0 + ty + 1) * HALOW + (ox0 + tx + 1)) : pix0;
    const int s1 = in_halo ? ((oy1 + ty + 1) * HALOW + (ox1 + tx + 1)) : pix1;
    const unsigned short* ap0 = Ab + (size_t)s0 * Cin + rsel * 16;  // A: contiguous K-chunk of 16
    const unsigned short* ap1 = Ab + (size_t)s1 * Cin + rsel * 16;
    const unsigned short* wp0 = Wb + ((size_t)t * CoutPad + n0 + lmod) * Cin + rsel * 16;
    const unsigned short* wp1 = wp0 + (size_t)16 * Cin;
    for (int k = 0; k < Cin; k += 32) {
      ABFrag a0, a1;
      a0.h[0] = *(const v8bf*)(ap0 + k); a0.h[1] = *(const v8bf*)(ap0 + k + 8);
      a1.h[0] = *(const v8bf*)(ap1 + k); a1.h[1] = *(const v8bf*)(ap1 + k + 8);
      v16bf b0 = *(const v16bf*)(wp0 + k);
      v16bf b1 = *(const v16bf*)(wp1 + k);
      __builtin_prefetch(wp0 + k + 64, 0, 0);   // global_prefetch_b8
      acc[0][0] = WMMA_BF16(a0.v, b0, acc[0][0]);
      acc[0][1] = WMMA_BF16(a0.v, b1, acc[0][1]);
      acc[1][0] = WMMA_BF16(a1.v, b0, acc[1][0]);
      acc[1][1] = WMMA_BF16(a1.v, b1, acc[1][1]);
    }
  }
  unsigned short* ob = out + (size_t)b * HALOPIX * CoutPad;
#pragma unroll
  for (int ni = 0; ni < 2; ++ni) {
    const int col = n0 + ni * 16 + lmod;
    const float bia = bias ? bias[col] : 0.0f;
#pragma unroll
    for (int mi = 0; mi < 2; ++mi) {
#pragma unroll
      for (int i = 0; i < 8; ++i) {
        int row = m0 + mi * 16 + rsel * 8 + i;
        int hy = row / RH + 1, hx = row % RH + 1;
        float v = acc[mi][ni][i] + bia;
        if (relu) v = fmaxf(v, 0.0f);
        ob[((size_t)hy * HALOW + hx) * CoutPad + col] = f2bf(v);
      }
    }
  }
}

// ---------------- fused gamma/beta 3x3 conv + LayerNorm modulation epilogue ------------------
// A = h bf16 halo layout [b][HALOPIX][Cin]. Wave computes 32(M)x16(N), gamma+beta share A.
// out[b][HW][Cout] (fp32) = xn * (1 + gamma) + beta
__global__ void wmma_conv_gb_apply(const unsigned short* __restrict__ A,
                                   const unsigned short* __restrict__ Wg,  // [9][CoutPad][Cin]
                                   const float* __restrict__ bg,
                                   const unsigned short* __restrict__ Wb_,
                                   const float* __restrict__ bb,
                                   const float* __restrict__ x,
                                   const float* __restrict__ stats,        // [b]{mean,var}
                                   float* __restrict__ out,
                                   int Cin, int Cout, int CoutPad, int x_channel_major) {
  const int lane = threadIdx.x & 31;
  const int wave = threadIdx.x >> 5;
  const int n0   = blockIdx.y * 64 + wave * 16;
  if (n0 >= CoutPad) return;
  const int m0   = blockIdx.x * 32;
  const int b    = blockIdx.z;
  const int rsel = lane >> 4;
  const int lmod = lane & 15;
  const int col  = n0 + lmod;
  const int pix0 = m0 + lmod, pix1 = pix0 + 16;
  const int oy0 = pix0 / RH, ox0 = pix0 % RH;
  const int oy1 = pix1 / RH, ox1 = pix1 % RH;
  const unsigned short* Ab = A + (size_t)b * HALOPIX * Cin;
  v8f ag[2], ab[2];
#pragma unroll
  for (int m = 0; m < 2; ++m)
#pragma unroll
    for (int i = 0; i < 8; ++i) { ag[m][i] = 0.0f; ab[m][i] = 0.0f; }

  for (int t = 0; t < 9; ++t) {
    const int ty = t / 3 - 1, tx = t % 3 - 1;
    const int s0 = (oy0 + ty + 1) * HALOW + (ox0 + tx + 1);
    const int s1 = (oy1 + ty + 1) * HALOW + (ox1 + tx + 1);
    const unsigned short* ap0 = Ab  + (size_t)s0 * Cin + rsel * 16;
    const unsigned short* ap1 = Ab  + (size_t)s1 * Cin + rsel * 16;
    const unsigned short* wgp = Wg  + ((size_t)t * CoutPad + col) * Cin + rsel * 16;
    const unsigned short* wbp = Wb_ + ((size_t)t * CoutPad + col) * Cin + rsel * 16;
    for (int k = 0; k < Cin; k += 32) {
      ABFrag a0, a1;
      a0.h[0] = *(const v8bf*)(ap0 + k); a0.h[1] = *(const v8bf*)(ap0 + k + 8);
      a1.h[0] = *(const v8bf*)(ap1 + k); a1.h[1] = *(const v8bf*)(ap1 + k + 8);
      v16bf gfrag = *(const v16bf*)(wgp + k);
      v16bf bfrag = *(const v16bf*)(wbp + k);
      __builtin_prefetch(wgp + k + 64, 0, 0);
      ag[0] = WMMA_BF16(a0.v, gfrag, ag[0]);
      ab[0] = WMMA_BF16(a0.v, bfrag, ab[0]);
      ag[1] = WMMA_BF16(a1.v, gfrag, ag[1]);
      ab[1] = WMMA_BF16(a1.v, bfrag, ab[1]);
    }
  }
  const float mean = stats[b * 2 + 0];
  const float rstd = rsqrtf(stats[b * 2 + 1] + LNEPS);
  const bool  cvalid = (col < Cout);
  const float biag = cvalid ? bg[col] : 0.0f;
  const float biab = cvalid ? bb[col] : 0.0f;
  float* ob = out + (size_t)b * HW * Cout;
  if (cvalid) {
#pragma unroll
    for (int mi = 0; mi < 2; ++mi) {
#pragma unroll
      for (int i = 0; i < 8; ++i) {
        int row = m0 + mi * 16 + rsel * 8 + i;
        float xv = x_channel_major ? x[((size_t)b * Cout + col) * HW + row]
                                   : x[((size_t)b * HW + row) * Cout + col];
        float xn = (xv - mean) * rstd;
        float g  = ag[mi][i] + biag;
        float be = ab[mi][i] + biab;
        ob[(size_t)row * Cout + col] = xn * (1.0f + g) + be;
      }
    }
  }
}

// ---------------- per-pixel 1x1 conv + softplus ----------------
__global__ void pointwise_softplus(const float* __restrict__ in, const float* __restrict__ w,
                                   const float* __restrict__ bias, float* __restrict__ out,
                                   int Cin, int Cout, int total) {
  int idx = blockIdx.x * blockDim.x + threadIdx.x;
  if (idx >= total) return;
  int co = idx % Cout;
  int bp = idx / Cout;                       // b*HW + pix
  const float* ip = in + (size_t)bp * Cin;
  const float* wp = w + (size_t)co * Cin;
  float s = bias[co];
#pragma unroll 8
  for (int i = 0; i < Cin; ++i) s += ip[i] * wp[i];
  out[idx] = fmaxf(s, 0.0f) + log1pf(expf(-fabsf(s)));   // stable softplus
}

// =====================================================================================
extern "C" void kernel_launch(void* const* d_in, const int* in_sizes, int n_in,
                              void* d_out, int out_size, void* d_ws, size_t ws_size,
                              hipStream_t stream) {
  const float* x_main = (const float*)d_in[0];
  const float* f_sem  = (const float*)d_in[1];
  const int*   segmap = (const int*)d_in[2];
  const float* wsA[3] = {(const float*)d_in[3],  (const float*)d_in[9],  (const float*)d_in[15]};
  const float* bsA[3] = {(const float*)d_in[4],  (const float*)d_in[10], (const float*)d_in[16]};
  const float* wgA[3] = {(const float*)d_in[5],  (const float*)d_in[11], (const float*)d_in[17]};
  const float* bgA[3] = {(const float*)d_in[6],  (const float*)d_in[12], (const float*)d_in[18]};
  const float* wbA[3] = {(const float*)d_in[7],  (const float*)d_in[13], (const float*)d_in[19]};
  const float* bbA[3] = {(const float*)d_in[8],  (const float*)d_in[14], (const float*)d_in[20]};
  const float* w0 = (const float*)d_in[21]; const float* bias0 = (const float*)d_in[22];
  const float* w1 = (const float*)d_in[23]; const float* bias1 = (const float*)d_in[24];
  const float* w2 = (const float*)d_in[25]; const float* bias2 = (const float*)d_in[26];
  float* outp = (float*)d_out;

  // -------- workspace carve-out (256B aligned) --------
  char* base = (char*)d_ws;
  size_t off = 0;
  auto alloc = [&](size_t bytes) -> void* {
    void* p = base + off;
    off = (off + bytes + 255) & ~(size_t)255;
    return p;
  };
  unsigned short* meansT = (unsigned short*)alloc((size_t)BATCH * CD * SEGS * 2);
  unsigned short* wseg   = (unsigned short*)alloc((size_t)BATCH * HW * SEGS * 2);
  unsigned short* sem    = (unsigned short*)alloc((size_t)BATCH * HALOPIX * CD * 2);   // halo
  unsigned short* hconv  = (unsigned short*)alloc((size_t)BATCH * HALOPIX * HID * 2);  // halo
  unsigned short *pk_ws[3], *pk_wg[3], *pk_wb[3];
  for (int l = 0; l < 3; ++l) pk_ws[l] = (unsigned short*)alloc((size_t)9 * HID * CD * 2);
  pk_wg[0] = (unsigned short*)alloc((size_t)9 * CM * HID * 2);
  pk_wb[0] = (unsigned short*)alloc((size_t)9 * CM * HID * 2);
  pk_wg[1] = (unsigned short*)alloc((size_t)9 * 16 * HID * 2);
  pk_wb[1] = (unsigned short*)alloc((size_t)9 * 16 * HID * 2);
  pk_wg[2] = (unsigned short*)alloc((size_t)9 * 16 * HID * 2);
  pk_wb[2] = (unsigned short*)alloc((size_t)9 * 16 * HID * 2);
  float* part   = (float*)alloc((size_t)BATCH * 64 * 2 * 4);
  float* stats0 = (float*)alloc(256);
  float* stats1 = (float*)alloc(256);
  float* stats2 = (float*)alloc(256);
  float* h1     = (float*)alloc((size_t)BATCH * HW * 8 * 4);
  float* h2     = (float*)alloc((size_t)BATCH * HW * 16 * 4);
  float* spade  = (float*)alloc((size_t)BATCH * HW * CM * 4);   // reused for all 3 layers
  (void)ws_size; (void)n_in; (void)in_sizes; (void)out_size;

  // halo buffers start at zero every call (interiors fully rewritten below)
  {
    int nsem = BATCH * HALOPIX * CD / 2;    // u32 words
    int nh   = BATCH * HALOPIX * HID / 2;
    zero_u32<<<(nsem + 255) / 256, 256, 0, stream>>>((unsigned int*)sem, nsem);
    zero_u32<<<(nh + 255) / 256, 256, 0, stream>>>((unsigned int*)hconv, nh);
  }

  auto repack = [&](const float* w, unsigned short* o, int O, int I, int Opad) {
    int total = 9 * Opad * I;
    repack_w<<<(total + 255) / 256, 256, 0, stream>>>(w, o, O, I, Opad, total);
  };
  for (int l = 0; l < 3; ++l) repack(wsA[l], pk_ws[l], HID, CD, HID);
  repack(wgA[0], pk_wg[0], CM, HID, CM);
  repack(wbA[0], pk_wb[0], CM, HID, CM);
  repack(wgA[1], pk_wg[1], 8,  HID, 16);
  repack(wbA[1], pk_wb[1], 8,  HID, 16);
  repack(wgA[2], pk_wg[2], 16, HID, 16);
  repack(wbA[2], pk_wb[2], 16, HID, 16);

  // segment means + resize weights
  compute_meansT<<<BATCH, 128, 0, stream>>>(f_sem, segmap, meansT);
  compute_wseg<<<BATCH * 36, 64, 0, stream>>>(segmap, wseg);

  // sem = Wseg (2304x64) @ meansT^T (64x768) -> bf16 halo [b][HALOPIX][CD]
  wmma_conv_bf16<<<dim3(HW / 32, CD / 128, BATCH), 128, 0, stream>>>(
      wseg, meansT, nullptr, sem, SEGS, CD, 1, (long long)CD * SEGS, 0, /*in_halo=*/0);

  // LN stats of x_main
  stats_partial<<<BATCH * 64, 256, 0, stream>>>(x_main, part, CM * HW, 64);
  stats_final<<<BATCH, 64, 0, stream>>>(part, stats0, 64, 1.0f / (float)(CM * HW));

  // ---- layer 0 ----
  wmma_conv_bf16<<<dim3(HW / 32, 1, BATCH), 128, 0, stream>>>(
      sem, pk_ws[0], bsA[0], hconv, CD, HID, 9, 0, 1, 1);
  wmma_conv_gb_apply<<<dim3(HW / 32, CM / 64, BATCH), 128, 0, stream>>>(
      hconv, pk_wg[0], bgA[0], pk_wb[0], bbA[0], x_main, stats0, spade, HID, CM, CM, 1);
  pointwise_softplus<<<(BATCH * HW * 8 + 255) / 256, 256, 0, stream>>>(
      spade, w0, bias0, h1, CM, 8, BATCH * HW * 8);
  stats_partial<<<BATCH * 8, 256, 0, stream>>>(h1, part, HW * 8, 8);
  stats_final<<<BATCH, 64, 0, stream>>>(part, stats1, 8, 1.0f / (float)(HW * 8));

  // ---- layer 1 ----
  wmma_conv_bf16<<<dim3(HW / 32, 1, BATCH), 128, 0, stream>>>(
      sem, pk_ws[1], bsA[1], hconv, CD, HID, 9, 0, 1, 1);
  wmma_conv_gb_apply<<<dim3(HW / 32, 1, BATCH), 128, 0, stream>>>(
      hconv, pk_wg[1], bgA[1], pk_wb[1], bbA[1], h1, stats1, spade, HID, 8, 16, 0);
  pointwise_softplus<<<(BATCH * HW * 16 + 255) / 256, 256, 0, stream>>>(
      spade, w1, bias1, h2, 8, 16, BATCH * HW * 16);
  stats_partial<<<BATCH * 8, 256, 0, stream>>>(h2, part, HW * 16, 8);
  stats_final<<<BATCH, 64, 0, stream>>>(part, stats2, 8, 1.0f / (float)(HW * 16));

  // ---- layer 2 ----
  wmma_conv_bf16<<<dim3(HW / 32, 1, BATCH), 128, 0, stream>>>(
      sem, pk_ws[2], bsA[2], hconv, CD, HID, 9, 0, 1, 1);
  wmma_conv_gb_apply<<<dim3(HW / 32, 1, BATCH), 128, 0, stream>>>(
      hconv, pk_wg[2], bgA[2], pk_wb[2], bbA[2], h2, stats2, spade, HID, 16, 16, 0);
  pointwise_softplus<<<(BATCH * HW * 1 + 255) / 256, 256, 0, stream>>>(
      spade, w2, bias2, outp, 16, 1, BATCH * HW);
}
